// CAM_28587302322763
// MI455X (gfx1250) — matrix operand order repaired and checked by reading, not compiled, not run on hardware
//
#include <hip/hip_runtime.h>

// Fused channel-attention (flash-style) for MI455X / gfx1250, wave32 + WMMA bf16.
// B=16, C=1024, HW=1024. One workgroup = 256 threads (8 waves) handles one
// (batch, 16-row query block). S and P stay in LDS; O stays in registers.
// f32 -> bf16 via single v_perm_b32 per pair (truncation).

#define CC 1024   // channels (attention sequence length)
#define HWD 1024  // spatial dim (head dim)

typedef __attribute__((ext_vector_type(16))) __bf16 v16bf;
typedef __attribute__((ext_vector_type(8)))  float  v8f;

union Frag {
    v16bf v;
    unsigned short us[16];
    unsigned int   ui[8];
    uint4          q[2];
};

// f32 -> bf16 scalar, truncating (top 16 bits): 1 VALU op.
__device__ __forceinline__ unsigned short f2bf(float x) {
    return (unsigned short)(__float_as_uint(x) >> 16);
}

// Pack two f32 into one dword of bf16x2 (lo -> [15:0], hi -> [31:16]).
// Single v_perm_b32: bytes {7,6} of hi and {3,2} of lo.
__device__ __forceinline__ unsigned int pk2bf(float lo, float hi) {
#if __has_builtin(__builtin_amdgcn_perm)
    return __builtin_amdgcn_perm(__float_as_uint(hi), __float_as_uint(lo), 0x07060302u);
#else
    return (__float_as_uint(lo) >> 16) | (__float_as_uint(hi) & 0xffff0000u);
#endif
}

// Load a 16-half WMMA fragment from LDS: 8 halves at p, 8 halves at p+off2.
// off2 = 8 for A-fragments too (contiguous K..K+15 pattern),
// off2 = 8  for contiguous-16 B-fragments.
__device__ __forceinline__ v16bf ld_frag_lds(const unsigned short* p, int off2) {
    Frag f;
    f.q[0] = *reinterpret_cast<const uint4*>(p);
    f.q[1] = *reinterpret_cast<const uint4*>(p + off2);
    return f.v;
}

// Load 16 contiguous f32 from global, convert to a bf16 B-fragment (8x v_perm).
__device__ __forceinline__ v16bf ld_frag_gl(const float* p) {
    const float4* p4 = reinterpret_cast<const float4*>(p);
    float4 x0 = p4[0], x1 = p4[1], x2 = p4[2], x3 = p4[3];
    Frag f;
    f.ui[0] = pk2bf(x0.x, x0.y); f.ui[1] = pk2bf(x0.z, x0.w);
    f.ui[2] = pk2bf(x1.x, x1.y); f.ui[3] = pk2bf(x1.z, x1.w);
    f.ui[4] = pk2bf(x2.x, x2.y); f.ui[5] = pk2bf(x2.z, x2.w);
    f.ui[6] = pk2bf(x3.x, x3.y); f.ui[7] = pk2bf(x3.z, x3.w);
    return f.v;
}

// LDS layout (dynamic):
//  [0, 32768)           : Q block as bf16 [16][1024]; reused as P after softmax
//  [32768, 32768+81920) : S as f32 [16][1024] (phases 1-2), then Vt bf16 [1024][40] (phase 3)
//  then: red[256] f32, rowmax[16] f32, rowinv[16] f32
#define LDS_R1   32768
#define LDS_RED  (32768 + 81920)
#define LDS_TOTAL (LDS_RED + (256 + 16 + 16) * 4)
#define VT_STRIDE 40   // halves; 80B rows -> 16B aligned, conflict-free frag reads

__global__ __launch_bounds__(256) void cam_fused_kernel(
    const float* __restrict__ Ares,   // a: V matrix and residual
    const float* __restrict__ Qg,     // b: query side of scores
    const float* __restrict__ Kg,     // c: key side of scores
    float* __restrict__ Og)
{
    extern __shared__ char lds[];
    unsigned short* qp  = reinterpret_cast<unsigned short*>(lds);            // Q then P
    float*          S   = reinterpret_cast<float*>(lds + LDS_R1);            // scores
    unsigned short* vt  = reinterpret_cast<unsigned short*>(lds + LDS_R1);   // V transposed
    float*          red = reinterpret_cast<float*>(lds + LDS_RED);
    float*          rowmax = red + 256;
    float*          rowinv = rowmax + 16;

    const int tid   = threadIdx.x;
    const int lane  = tid & 31;
    const int wave  = tid >> 5;
    const int wg    = blockIdx.x;
    const int batch = wg >> 6;          // 64 row-blocks per batch
    const int m0    = (wg & 63) << 4;   // first query row of this block

    const size_t bstride = (size_t)CC * HWD;
    const float* Qb = Qg   + (size_t)batch * bstride;
    const float* Kb = Kg   + (size_t)batch * bstride;
    const float* Vb = Ares + (size_t)batch * bstride;
    float*       Ob = Og   + (size_t)batch * bstride;

    // Fragment geometry (per ISA VGPR layout tables)
    const int arow  = lane & 15;          // A row / B column / D column index
    const int ahalf = (lane >> 4) << 3;
    const int bksel = (lane >> 4) << 4;   // B-frag: K base (0 or 16)
    const int rbase = (lane >> 4) << 3;   // D: M = rbase + vgpr

    // ---------------- phase 0: stage Q rows [m0, m0+16) as bf16 ----------------
    {
        const float4* src = reinterpret_cast<const float4*>(Qb + (size_t)m0 * HWD);
        #pragma unroll
        for (int i = 0; i < 16; ++i) {
            int v = i * 256 + tid;          // float4 index within 16x1024
            float4 t = src[v];
            uint2 u;
            u.x = pk2bf(t.x, t.y);
            u.y = pk2bf(t.z, t.w);
            *reinterpret_cast<uint2*>(qp + 4 * v) = u;
        }
    }
    __syncthreads();

    // ---------------- phase 1: S = Q * K^T (each wave: 16x128 slice) ------------
    {
        const int jt0 = wave << 7;
        for (int t = 0; t < 8; ++t) {
            const int jt = jt0 + (t << 4);
            // B operand of Q*K^T: element (k, j) = K[j][k]  -> contiguous in k
            const float* krow = Kb + (size_t)(jt + arow) * HWD + bksel;
            v8f acc = {};
            #pragma unroll 4
            for (int kb = 0; kb < HWD; kb += 32) {
                v16bf af = ld_frag_lds(qp + arow * HWD + kb + bksel, 8);
                v16bf bf = ld_frag_gl(krow + kb);
                acc = __builtin_amdgcn_wmma_f32_16x16x32_bf16(
                        false, af, false, bf, (short)0, acc, false, false);
            }
            const int col = jt + arow;
            #pragma unroll
            for (int r = 0; r < 8; ++r)
                S[(rbase + r) * CC + col] = acc[r];
        }
    }
    __syncthreads();

    // ---------------- phase 2: softmax over rows of S; P = exp(S-m) as bf16 -----
    {
        const int row = tid >> 4;
        const int sub = tid & 15;
        float m = -3.402823e38f;
        for (int j = sub; j < CC; j += 16)
            m = fmaxf(m, S[row * CC + j]);
        red[tid] = m;
        __syncthreads();
        if (sub == 0) {
            float mm = red[row << 4];
            #pragma unroll
            for (int k2 = 1; k2 < 16; ++k2) mm = fmaxf(mm, red[(row << 4) + k2]);
            rowmax[row] = mm;
        }
        __syncthreads();
        const float mm = rowmax[row];
        float s = 0.f;
        for (int j = sub; j < CC; j += 16) {
            float e = __expf(S[row * CC + j] - mm);
            s += e;
            qp[row * CC + j] = f2bf(e);     // P overwrites Q region
        }
        __syncthreads();                    // red free again; S reads done
        red[tid] = s;
        __syncthreads();
        if (sub == 0) {
            float ss = 0.f;
            #pragma unroll
            for (int k2 = 0; k2 < 16; ++k2) ss += red[(row << 4) + k2];
            rowinv[row] = 1.0f / ss;
        }
    }
    __syncthreads();

    // ---------------- phase 3: O = P * V (wave owns 16x128 of O in regs) --------
    v8f accO[8];
    #pragma unroll
    for (int t = 0; t < 8; ++t) accO[t] = (v8f){};

    const int nt0 = wave << 7;
    for (int jb = 0; jb < CC; jb += 32) {
        __syncthreads();   // prior vt reads (or S use on first iter) complete
        // Cooperative transpose of V rows [jb, jb+32) into vt[n][j] bf16
        const float4* src = reinterpret_cast<const float4*>(Vb + (size_t)jb * HWD);
        #pragma unroll
        for (int i = 0; i < 32; ++i) {
            int v = i * 256 + tid;
            float4 tt = src[v];
            int e = v * 4;
            int j = e >> 10;       // 0..31
            int n = e & 1023;
            vt[(n + 0) * VT_STRIDE + j] = f2bf(tt.x);
            vt[(n + 1) * VT_STRIDE + j] = f2bf(tt.y);
            vt[(n + 2) * VT_STRIDE + j] = f2bf(tt.z);
            vt[(n + 3) * VT_STRIDE + j] = f2bf(tt.w);
        }
        __syncthreads();
        // A fragment (P) shared across the 8 n-tiles of this wave
        v16bf pf = ld_frag_lds(qp + arow * CC + jb + bksel, 8);
        #pragma unroll
        for (int t = 0; t < 8; ++t) {
            const unsigned short* vb = vt + (nt0 + (t << 4) + arow) * VT_STRIDE + bksel;
            v16bf vf = ld_frag_lds(vb, 8);
            accO[t] = __builtin_amdgcn_wmma_f32_16x16x32_bf16(
                        false, pf, false, vf, (short)0, accO[t], false, false);
        }
    }

    // ---------------- epilogue: normalize, residual, store ----------------------
    {
        #pragma unroll
        for (int t = 0; t < 8; ++t) {
            const int n = nt0 + (t << 4) + arow;
            #pragma unroll
            for (int r = 0; r < 8; ++r) {
                const int m = rbase + r;
                const size_t gi = (size_t)(m0 + m) * HWD + n;
                Ob[gi] = accO[t][r] * rowinv[m] + Vb[gi];
            }
        }
    }
}

extern "C" void kernel_launch(void* const* d_in, const int* in_sizes, int n_in,
                              void* d_out, int out_size, void* d_ws, size_t ws_size,
                              hipStream_t stream) {
    (void)in_sizes; (void)n_in; (void)d_ws; (void)ws_size; (void)out_size;
    const float* a = (const float*)d_in[0];   // V + residual
    const float* b = (const float*)d_in[1];   // Q
    const float* c = (const float*)d_in[2];   // K
    float* out = (float*)d_out;

    dim3 grid(16 * 64);   // 16 batches x 64 query row-blocks
    dim3 block(256);      // 8 waves (wave32)
    cam_fused_kernel<<<grid, block, LDS_TOTAL, stream>>>(a, b, c, out);
}
